// ODENet_3178275799407
// MI455X (gfx1250) — matrix operand order repaired and checked by reading, compiled, not run on hardware
//
#include <hip/hip_runtime.h>

// ODENet scan: S=2048 steps, B=512 independent chains, H=128.
// 32 blocks x 32 threads (1 wave); each wave owns 16 batch rows for all 2048 steps.
// Layer 2 (the only big matmul) runs on v_wmma_f32_16x16x32_f16 as h2^T = W2^T * h1^T.

typedef __attribute__((ext_vector_type(16))) _Float16 v16h;
typedef __attribute__((ext_vector_type(8)))  float    v8f;

#define S_LEN 2048
#define B_LEN 512
#define H_LEN 128

__device__ __forceinline__ float fast_tanh(float x) {
#if __has_builtin(__builtin_amdgcn_tanhf)
  return __builtin_amdgcn_tanhf(x);           // CDNA5 v_tanh_f32 (TRANS op)
#else
  // tanh(x) = 1 - 2/(exp(2x)+1);  exp(2x) = exp2(x * 2*log2(e))
  float e = __builtin_amdgcn_exp2f(x * 2.8853900817779268f);
  return 1.0f - 2.0f * __builtin_amdgcn_rcpf(e + 1.0f);
#endif
}

__global__ __launch_bounds__(32, 1)
void odenet_scan(const float* __restrict__ X,   // (S, B, 1)
                 const float* __restrict__ W1,  // (2, H)
                 const float* __restrict__ B1,  // (H)
                 const float* __restrict__ W2,  // (H, H)
                 const float* __restrict__ B2,  // (H)
                 const float* __restrict__ W3,  // (H, 2)
                 const float* __restrict__ B3,  // (2)
                 float* __restrict__ Out)       // (S, B, 1)
{
  // W2^T pre-swizzled into WMMA 16-bit A-matrix fragment layout.
  // Fragment (t = n-tile 0..7, c = k-chunk 0..3), lane L holds 16 f16:
  //   element e -> A[M'][K] with M' = n = 16t + (L&15),
  //   K = 32c + 16*(L>>4) + e   (contiguous k order, matches B fragment)
  __shared__ v16h w2frag[8][4][32];            // 32 KB LDS

  const int lane = threadIdx.x;                // 0..31 (wave32)
  const int hi   = lane >> 4;
  const int m    = lane & 15;                  // batch row within tile
  const int b0   = blockIdx.x * 16;

  // ---------------- one-time setup ----------------
  for (int t = 0; t < 8; ++t)
    for (int c = 0; c < 4; ++c) {
      v16h f;
      #pragma unroll
      for (int e = 0; e < 16; ++e) {
        int k  = 32*c + 16*hi + e;                      // K index into W2 rows
        int nn = 16*t + m;                              // output feature n
        f[e] = (_Float16)W2[k*H_LEN + nn];              // (W2^T)[n][k]
      }
      w2frag[t][c][lane] = f;
    }
  __syncthreads();   // single-wave WG: lowers to s_nop + waits, cheap

  // Per-lane loop-invariant constants (registers).
  // Lane (hi,m) produces h1[m][k] for k = 32c + 16*hi + j  (j=0..15) — exactly
  // the contiguous k order of the w2frag A fragments, so no shuffle is ever needed.
  float w10[64], w11[64], bb1[64];
  #pragma unroll
  for (int c = 0; c < 4; ++c)
    #pragma unroll
    for (int j = 0; j < 16; ++j) {
      int k = 32*c + 16*hi + j;
      w10[c*16+j] = W1[k];            // W1[0][k]
      w11[c*16+j] = W1[H_LEN + k];    // W1[1][k]
      bb1[c*16+j] = B1[k];
    }
  // f32 D-layout: lane holds (n = 16t + 8*hi + r, m = lane&15) in acc[r].
  float b2v[64], w3v[64];
  #pragma unroll
  for (int t = 0; t < 8; ++t)
    #pragma unroll
    for (int r = 0; r < 8; ++r) {
      int nn = 16*t + 8*hi + r;
      b2v[t*8+r] = B2[nn];
      w3v[t*8+r] = W3[2*nn + 1];      // only column 1 of W3 is ever used
    }
  const float b3s = B3[1];

  // ---------------- sequential scan ----------------
  float s  = 0.0f;                    // state for batch row m (uniform across hi halves)
  float xv = X[b0 + m];

  for (int n = 0; n < S_LEN; ++n) {
    // prefetch next step's x so the global load overlaps this step's MLP
    int   np    = (n + 1 < S_LEN) ? (n + 1) : n;
    float xnext = X[np*B_LEN + b0 + m];

    // layer 1: h1 = tanh(x*W1[0] + s*W1[1] + b1), built directly as B-fragments
    v16h hf[4];
    #pragma unroll
    for (int c = 0; c < 4; ++c)
      #pragma unroll
      for (int j = 0; j < 16; ++j) {
        float pre = fmaf(xv, w10[c*16+j], fmaf(s, w11[c*16+j], bb1[c*16+j]));
        hf[c][j] = (_Float16)fast_tanh(pre);
      }

    // layers 2+3: h2^T tiles via WMMA, then dot with W3[:,1]
    float part = 0.0f;
    #pragma unroll
    for (int t = 0; t < 8; ++t) {
      v8f acc;
      #pragma unroll
      for (int r = 0; r < 8; ++r) acc[r] = b2v[t*8+r];
      #pragma unroll
      for (int c = 0; c < 4; ++c)
        acc = __builtin_amdgcn_wmma_f32_16x16x32_f16(
                  false, w2frag[t][c][lane],   // A = W2^T fragment (from LDS)
                  false, hf[c],                // B = h1^T fragment (registers)
                  (short)0, acc, false, false);
      #pragma unroll
      for (int r = 0; r < 8; ++r)
        part = fmaf(fast_tanh(acc[r]), w3v[t*8+r], part);
    }
    // reduce the two half-wave partials (n-ranges) -> full y[m] in every lane
    part += __shfl_xor(part, 16);

    s = s + (part + b3s);             // dt = 1.0
    if (lane < 16) Out[n*B_LEN + b0 + lane] = s;

    xv = xnext;
  }
}

extern "C" void kernel_launch(void* const* d_in, const int* in_sizes, int n_in,
                              void* d_out, int out_size, void* d_ws, size_t ws_size,
                              hipStream_t stream) {
  (void)in_sizes; (void)n_in; (void)d_ws; (void)ws_size; (void)out_size;
  const float* X  = (const float*)d_in[0];
  const float* W1 = (const float*)d_in[1];
  const float* B1 = (const float*)d_in[2];
  const float* W2 = (const float*)d_in[3];
  const float* B2 = (const float*)d_in[4];
  const float* W3 = (const float*)d_in[5];
  const float* B3 = (const float*)d_in[6];
  odenet_scan<<<dim3(B_LEN/16), dim3(32), 0, stream>>>(X, W1, B1, W2, B2, W3, B3,
                                                       (float*)d_out);
}
